// MiniSelfAttention_21071109554380
// MI455X (gfx1250) — matrix operand order repaired and checked by reading, not compiled, not run on hardware
//
#include <hip/hip_runtime.h>
#include <hip/hip_bf16.h>

// ---------------------------------------------------------------------------
// MiniSelfAttention for MI455X (gfx1250), wave32 + v_wmma_f32_16x16x32_bf16.
// Pipeline: [Q,K,V = x@W (bf16)] -> [P = softmax(Q K^T) (bf16)]
//           -> [O = P@V (bf16)] -> [out = O@Wp + bp (f32)]
// Workspace (bf16): Q,K,V (3 x 32MB), P (64MB), O (32MB) ~= 201 MB.
// ---------------------------------------------------------------------------

typedef __attribute__((ext_vector_type(16))) __bf16 v16bf;
typedef __attribute__((ext_vector_type(8)))  float  v8f;
typedef int v4i __attribute__((vector_size(16)));   // matches builtin param type

// ---- CDNA5 async global->LDS path (probe-confirmed present) ---------------
#if defined(__has_builtin)
#if __has_builtin(__builtin_amdgcn_global_load_async_to_lds_b128) && \
    __has_builtin(__builtin_amdgcn_s_wait_asynccnt)
#define HAVE_ASYNC_LDS 1
#endif
#endif
#ifndef HAVE_ASYNC_LDS
#define HAVE_ASYNC_LDS 0
#endif

#if HAVE_ASYNC_LDS
#define AS1 __attribute__((address_space(1)))
#define AS3 __attribute__((address_space(3)))
__device__ __forceinline__ void async_cp16(void* lds_dst, const void* gsrc) {
  // generic->AS casts via integers: generic shared ptr low 32 bits == LDS offset.
  __builtin_amdgcn_global_load_async_to_lds_b128(
      (AS1 v4i*)(unsigned long long)(__SIZE_TYPE__)gsrc,
      (AS3 v4i*)(unsigned)(__SIZE_TYPE__)lds_dst, 0, 0);
}
__device__ __forceinline__ void async_wait_all() {
  __builtin_amdgcn_s_wait_asynccnt(0);
}
#else
__device__ __forceinline__ void async_cp16(void* lds_dst, const void* gsrc) {
  *(uint4*)lds_dst = *(const uint4*)gsrc;
}
__device__ __forceinline__ void async_wait_all() {}
#endif

__device__ __forceinline__ unsigned short f2bf(float f) {
  unsigned u = __float_as_uint(f);
  u += 0x7FFFu + ((u >> 16) & 1u);            // round-to-nearest-even
  return (unsigned short)(u >> 16);
}
__device__ __forceinline__ unsigned short to_bf(float f)          { return f2bf(f); }
__device__ __forceinline__ unsigned short to_bf(unsigned short s) { return s; }

template <typename T>
struct alignas(4 * sizeof(T)) Vec4 { T x, y, z, w; };

union FragU { v16bf v; uint4 q[2]; };

__device__ __forceinline__ v8f zero8() {
  v8f z = {0.f, 0.f, 0.f, 0.f, 0.f, 0.f, 0.f, 0.f};
  return z;
}

// A fragment: 16x32 bf16, row-major source, row stride `ld` (elements).
// ISA: lanes 0-15 (row=lane): K 0-7 / 16-23; lanes 16-31: K 8-15 / 24-31.
__device__ __forceinline__ v16bf load_a_frag(const unsigned short* base, int ld, int lane) {
  int m  = lane & 15;
  int kb = (lane & 16) ? 8 : 0;
  const unsigned short* p = base + (size_t)m * ld;
  FragU f;
  f.q[0] = *(const uint4*)(p + kb);
  f.q[1] = *(const uint4*)(p + 16 + kb);
  return f.v;
}

// B fragment: 32x16 bf16 from N-major source (row n holds K contiguous, stride ld).
// ISA: lanes 0-15 (col=lane): K 0-7 / 16-23; lanes 16-31: K 8-15 / 24-31.
__device__ __forceinline__ v16bf load_bt_frag(const unsigned short* base, int ld, int lane) {
  int n  = lane & 15;
  int kb = (lane & 16) ? 8 : 0;
  const unsigned short* p = base + (size_t)n * ld + kb;
  FragU f;
  f.q[0] = *(const uint4*)(p);
  f.q[1] = *(const uint4*)(p + 16);
  return f.v;
}

__device__ __forceinline__ v8f wmma_bf16(v16bf a, v16bf b, v8f c) {
  return __builtin_amdgcn_wmma_f32_16x16x32_bf16(false, a, false, b, (short)0, c,
                                                 false, false);
}

#define LSTR 40   // LDS row stride (bf16): 80B rows -> 16B-aligned b128 accesses

// ---------------------------------------------------------------------------
// Tiled GEMM: C[M,N] = scale*(A@B) (+bias). Block 256 thr = 8 waves.
// Block tile 256(M) x 128(N); wave tile 32 x 128 (16 WMMAs / 32-wide K-step).
// DOUBLE-BUFFERED LDS (60 KB): next tile's async global->LDS b128 copies (bf16
// A) / f32 conversions are issued before computing the current tile, waited
// only after the 16 WMMAs.  B fragments use an explicit two-buffer register
// pipeline (unroll-by-2) so DS waits cover loads issued a full WMMA earlier.
// ---------------------------------------------------------------------------
template <typename AT, typename BT, bool OBF, bool BIAS>
__global__ __launch_bounds__(256) void gemm_wmma(
    const AT* __restrict__ A, const BT* __restrict__ Bm, void* __restrict__ Cp,
    const float* __restrict__ bias, int M, int N, int K,
    int lda, int ldb, int ldc, float scale,
    long long sA, long long sB, long long sC) {
  (void)M; (void)N;
  __shared__ unsigned short lA[2][256 * LSTR];   // 2 x 20 KB
  __shared__ unsigned short lB[2][128 * LSTR];   // 2 x 10 KB ([n][k] transposed)

  const int tid  = threadIdx.x;
  const int lane = tid & 31;
  const int wave = tid >> 5;
  const int m0   = blockIdx.y * 256;
  const int n0   = blockIdx.x * 128;

  const AT* Ab = A  + (size_t)blockIdx.z * (size_t)sA;
  const BT* Bb = Bm + (size_t)blockIdx.z * (size_t)sB;

  auto stageA = [&](unsigned short* dA, int kk) {
    if (sizeof(AT) == 2) {            // bf16 source: raw async b128 copies
      const unsigned short* As = (const unsigned short*)(const void*)Ab;
      #pragma unroll
      for (int i = 0; i < 4; ++i) {
        int e = tid + i * 256;        // 256 rows x 4 segments
        int r = e >> 2;
        int c = (e & 3) << 3;         // 0,8,16,24
        async_cp16(&dA[r * LSTR + c], As + (size_t)(m0 + r) * lda + kk + c);
      }
    } else {                           // f32 source: convert in-flight
      #pragma unroll
      for (int i = 0; i < 8; ++i) {
        int e = (tid + i * 256) << 2;  // 8192 elems
        int r = e >> 5;
        int c = e & 31;
        Vec4<AT> va = *(const Vec4<AT>*)(Ab + (size_t)(m0 + r) * lda + kk + c);
        unsigned short* d = &dA[r * LSTR + c];
        d[0] = to_bf(va.x); d[1] = to_bf(va.y); d[2] = to_bf(va.z); d[3] = to_bf(va.w);
      }
    }
  };
  auto stageB = [&](unsigned short* dB, int kk) {
    #pragma unroll
    for (int i = 0; i < 4; ++i) {
      int e  = (tid + i * 256) << 2;
      int k2 = e >> 7;
      int c  = e & 127;
      Vec4<BT> vb = *(const Vec4<BT>*)(Bb + (size_t)(kk + k2) * ldb + n0 + c);
      dB[(c + 0) * LSTR + k2] = to_bf(vb.x);
      dB[(c + 1) * LSTR + k2] = to_bf(vb.y);
      dB[(c + 2) * LSTR + k2] = to_bf(vb.z);
      dB[(c + 3) * LSTR + k2] = to_bf(vb.w);
    }
  };

  v8f acc[2][8];
  #pragma unroll
  for (int g = 0; g < 2; ++g)
    #pragma unroll
    for (int j = 0; j < 8; ++j) acc[g][j] = zero8();

  // Prologue: stage tile 0 into buffer 0.
  stageA(lA[0], 0);
  stageB(lB[0], 0);
  if (sizeof(AT) == 2) async_wait_all();
  __syncthreads();

  for (int k0 = 0; k0 < K; k0 += 32) {
    const int cur  = (k0 >> 5) & 1;
    const int nxt  = cur ^ 1;
    const bool more = (k0 + 32) < K;

    // Issue next tile's staging before computing on the current one.
    if (more) { stageA(lA[nxt], k0 + 32); stageB(lB[nxt], k0 + 32); }

    v16bf a0 = load_a_frag(&lA[cur][(wave * 32 + 0)  * LSTR], LSTR, lane);
    v16bf a1 = load_a_frag(&lA[cur][(wave * 32 + 16) * LSTR], LSTR, lane);
    // Two-buffer register pipeline over the 8 B fragments.
    v16bf b0 = load_bt_frag(&lB[cur][0 * 16 * LSTR], LSTR, lane);
    v16bf b1 = load_bt_frag(&lB[cur][1 * 16 * LSTR], LSTR, lane);
    #pragma unroll
    for (int j = 0; j < 8; j += 2) {
      acc[0][j] = wmma_bf16(a0, b0, acc[0][j]);
      acc[1][j] = wmma_bf16(a1, b0, acc[1][j]);
      if (j + 2 < 8) b0 = load_bt_frag(&lB[cur][(j + 2) * 16 * LSTR], LSTR, lane);
      acc[0][j + 1] = wmma_bf16(a0, b1, acc[0][j + 1]);
      acc[1][j + 1] = wmma_bf16(a1, b1, acc[1][j + 1]);
      if (j + 3 < 8) b1 = load_bt_frag(&lB[cur][(j + 3) * 16 * LSTR], LSTR, lane);
    }

    if (more && sizeof(AT) == 2) async_wait_all();
    __syncthreads();
  }

  // Writeback. C layout: VGPR r -> row r + 8*(lane>=16), col = lane&15.
  const int hi = (lane >> 4) & 1;
  const int nl = lane & 15;
  unsigned short* C16 = (unsigned short*)Cp + (size_t)blockIdx.z * (size_t)sC;
  float*          C32 = (float*)Cp          + (size_t)blockIdx.z * (size_t)sC;
  #pragma unroll
  for (int g = 0; g < 2; ++g) {
    #pragma unroll
    for (int j = 0; j < 8; ++j) {
      int n = n0 + j * 16 + nl;
      float bv = 0.0f;
      if (BIAS) bv = bias[n];
      #pragma unroll
      for (int r = 0; r < 8; ++r) {
        int row = m0 + wave * 32 + g * 16 + hi * 8 + r;
        float val = acc[g][j][r] * scale + bv;
        if (OBF) C16[(size_t)row * ldc + n] = f2bf(val);
        else     C32[(size_t)row * ldc + n] = val;
      }
    }
  }
}

// ---------------------------------------------------------------------------
// P[b] = softmax(Q[b] @ K[b]^T)  (softmax scale pre-folded into Q).
// Block 256 thr = 8 waves; wave owns 16 query rows; Q panel hoisted into 32
// v16bf fragments (256 VGPRs, CDNA5 big register file).  K streams through
// TWO ping-ponged 16x512 LDS half-chunk buffers filled by async global->LDS
// b128 copies (no VGPR transit => no spill pressure): while 16 WMMAs consume
// half h, the copies for half h+1 are in flight.  B fragments use an explicit
// two-buffer register pipeline (unroll-by-2).
// ---------------------------------------------------------------------------
#define HKW  512    // half-chunk width in d-elements
#define HPAD 520    // LDS row stride (1040B = 16B-aligned, bank-spread)

__global__ __launch_bounds__(256) void attn_softmax_kernel(
    const unsigned short* __restrict__ Q, const unsigned short* __restrict__ Kmat,
    unsigned short* __restrict__ P, int T, int D) {
  __shared__ unsigned short lK[2][16 * HPAD];   // 2 x 16.25 KB
  const int tid  = threadIdx.x;
  const int lane = tid & 31;
  const int wave = tid >> 5;
  const int b    = blockIdx.y;
  const unsigned short* Qb = Q    + (size_t)b * T * D;
  const unsigned short* Kb = Kmat + (size_t)b * T * D;
  unsigned short*       Pb = P    + (size_t)b * T * T;
  const int qrow0 = blockIdx.x * 128 + wave * 16;
  const unsigned short* Qbase = Qb + (size_t)qrow0 * D;
  const int hi = (lane >> 4) & 1;
  const int nl = lane & 15;

  // Async-copy one 16-key x 512 half-chunk into an LDS buffer.
  auto issue_half = [&](unsigned short* dst, const unsigned short* Kc, int dbase) {
    #pragma unroll
    for (int i = 0; i < 4; ++i) {
      int e   = tid + i * 256;       // 1024 x b128 segments
      int row = e >> 6;              // 0..15 (key)
      int col = (e & 63) << 3;       // 0..504 (d)
      async_cp16(&dst[row * HPAD + col], Kc + (size_t)row * D + dbase + col);
    }
  };
  // 16 WMMAs over one half-chunk; two-buffer register pipeline on B frags.
  auto half_tile = [&](const v16bf* qf16, const unsigned short* buf, v8f acc) {
    v16bf b0 = load_bt_frag(buf + 0 * 32, HPAD, lane);
    v16bf b1 = load_bt_frag(buf + 1 * 32, HPAD, lane);
    #pragma unroll
    for (int kd = 0; kd < 16; kd += 2) {
      acc = wmma_bf16(qf16[kd], b0, acc);
      if (kd + 2 < 16) b0 = load_bt_frag(buf + (kd + 2) * 32, HPAD, lane);
      acc = wmma_bf16(qf16[kd + 1], b1, acc);
      if (kd + 3 < 16) b1 = load_bt_frag(buf + (kd + 3) * 32, HPAD, lane);
    }
    return acc;
  };

  // Hoist Q panel: 32 fragments of 16x32.
  v16bf qf[32];
  #pragma unroll
  for (int kd = 0; kd < 32; ++kd) qf[kd] = load_a_frag(Qbase + kd * 32, D, lane);

  float m[8], l[8];
  #pragma unroll
  for (int r = 0; r < 8; ++r) { m[r] = -3.0e38f; l[r] = 0.0f; }

  // ---- pass 1: running row max / sum-exp ----
  issue_half(lK[0], Kb, 0);
  async_wait_all();
  __syncthreads();
  for (int kc = 0; kc < T; kc += 16) {          // cur buffer parity is 0 here
    const unsigned short* Kc = Kb + (size_t)kc * D;
    // half 0 (d 0..511): prefetch same chunk's upper half.
    issue_half(lK[1], Kc, HKW);
    v8f acc = half_tile(qf, lK[0], zero8());
    async_wait_all();
    __syncthreads();
    // half 1 (d 512..1023): prefetch next chunk's lower half.
    if (kc + 16 < T) issue_half(lK[0], Kc + (size_t)16 * D, 0);
    acc = half_tile(qf + 16, lK[1], acc);
    async_wait_all();
    __syncthreads();

    #pragma unroll
    for (int r = 0; r < 8; ++r) {
      float mx = acc[r];
      #pragma unroll
      for (int o = 1; o < 16; o <<= 1) mx = fmaxf(mx, __shfl_xor(mx, o, 32));
      float mn = fmaxf(m[r], mx);
      float e = __expf(acc[r] - mn);
      #pragma unroll
      for (int o = 1; o < 16; o <<= 1) e += __shfl_xor(e, o, 32);
      l[r] = l[r] * __expf(m[r] - mn) + e;
      m[r] = mn;
    }
  }
  float inv[8];
  #pragma unroll
  for (int r = 0; r < 8; ++r) inv[r] = 1.0f / l[r];

  // ---- pass 2: recompute S, emit normalized probabilities (bf16) ----
  issue_half(lK[0], Kb, 0);
  async_wait_all();
  __syncthreads();
  for (int kc = 0; kc < T; kc += 16) {
    const unsigned short* Kc = Kb + (size_t)kc * D;
    issue_half(lK[1], Kc, HKW);
    v8f acc = half_tile(qf, lK[0], zero8());
    async_wait_all();
    __syncthreads();
    if (kc + 16 < T) issue_half(lK[0], Kc + (size_t)16 * D, 0);
    acc = half_tile(qf + 16, lK[1], acc);
    async_wait_all();
    __syncthreads();

    #pragma unroll
    for (int r = 0; r < 8; ++r) {
      float p = __expf(acc[r] - m[r]) * inv[r];
      Pb[(size_t)(qrow0 + hi * 8 + r) * T + kc + nl] = f2bf(p);
    }
  }
}

// ---------------------------------------------------------------------------
// Host launcher
// ---------------------------------------------------------------------------
extern "C" void kernel_launch(void* const* d_in, const int* in_sizes, int n_in,
                              void* d_out, int out_size, void* d_ws, size_t ws_size,
                              hipStream_t stream) {
  (void)in_sizes; (void)n_in; (void)out_size; (void)ws_size;
  const float* x  = (const float*)d_in[0];
  const float* Wq = (const float*)d_in[1];
  const float* Wk = (const float*)d_in[2];
  const float* Wv = (const float*)d_in[3];
  const float* Wp = (const float*)d_in[4];
  const float* bp = (const float*)d_in[5];

  const int B = 8, T = 2048, D = 1024;
  const int M = B * T;                       // 16384 flattened rows

  size_t nTD = (size_t)B * T * D;            // 16,777,216 elements
  unsigned short* Q  = (unsigned short*)d_ws;
  unsigned short* Kw = Q  + nTD;
  unsigned short* V  = Kw + nTD;
  unsigned short* P  = V  + nTD;             // B*T*T = 33,554,432 elements
  unsigned short* O  = P  + (size_t)B * T * T;

  dim3 blk(256, 1, 1);
  dim3 gProj(D / 128, M / 256, 1);           // (8, 64)
  const float qscale = 1.0f / sqrtf((float)D);

  // Q,K,V projections: x(f32) @ W(f32) -> bf16. Softmax scale folded into Q.
  gemm_wmma<float, float, true, false><<<gProj, blk, 0, stream>>>(
      x, Wq, (void*)Q, nullptr, M, D, D, D, D, D, qscale, 0, 0, 0);
  gemm_wmma<float, float, true, false><<<gProj, blk, 0, stream>>>(
      x, Wk, (void*)Kw, nullptr, M, D, D, D, D, D, 1.0f, 0, 0, 0);
  gemm_wmma<float, float, true, false><<<gProj, blk, 0, stream>>>(
      x, Wv, (void*)V, nullptr, M, D, D, D, D, D, 1.0f, 0, 0, 0);

  // Attention probabilities.
  dim3 gAttn(T / 128, B, 1);                 // (16, 8)
  attn_softmax_kernel<<<gAttn, blk, 0, stream>>>(Q, Kw, P, T, D);

  // O = P @ V  (batched bf16 GEMM; A staged via async global->LDS).
  dim3 gPV(D / 128, T / 256, B);             // (8, 8, 8)
  gemm_wmma<unsigned short, unsigned short, true, false><<<gPV, blk, 0, stream>>>(
      P, V, (void*)O, nullptr, T, D, T, T, D, D, 1.0f,
      (long long)T * T, (long long)T * D, (long long)T * D);

  // out = O @ Wp + bp  (f32 output).
  gemm_wmma<unsigned short, float, false, true><<<gProj, blk, 0, stream>>>(
      O, Wp, d_out, bp, M, D, D, D, D, D, 1.0f, 0, 0, 0);
}
